// interpretable_multi_head_attention_74337293959305
// MI455X (gfx1250) — matrix operand order repaired and checked by reading, not compiled, not run on hardware
//
#include <hip/hip_runtime.h>

typedef _Float16 half_t;
typedef __attribute__((ext_vector_type(16))) _Float16 v16h;
typedef __attribute__((ext_vector_type(8)))  _Float16 v8h;
typedef __attribute__((ext_vector_type(2)))  _Float16 h2;
typedef __attribute__((ext_vector_type(8)))  float    v8f;

#define BB   2
#define SS   2048
#define DD   512
#define HH   8
#define DKK  64
#define ROWS (BB * SS)          // 4096
#define SMEM_BYTES (16 * SS * 4 + 16 * SS * 2)   // 196608: f32 scores strip + f16 prob strip

// ---------------- WMMA fragment helpers (wave32, v_wmma_f32_16x16x32_f16) ----------------
// A: 16x32 f16, row-major source. Lane L holds row M=L%16; g=L/16.
//   half-elems 0..7  -> K = 8g + {0..7};  half-elems 8..15 -> K = 16 + 8g + {0..7}
__device__ __forceinline__ v16h load_a_frag(const half_t* M, int ld, int m_base, int k_base) {
  int lane = threadIdx.x & 31;
  int g = lane >> 4;
  const half_t* row = M + (size_t)(m_base + (lane & 15)) * ld + k_base + 8 * g;
  v16h a;
#pragma unroll
  for (int j = 0; j < 4; ++j) {
    h2 p = *reinterpret_cast<const h2*>(row + 2 * j);
    a[2 * j] = p.x; a[2 * j + 1] = p.y;
  }
#pragma unroll
  for (int j = 0; j < 4; ++j) {
    h2 p = *reinterpret_cast<const h2*>(row + 16 + 2 * j);
    a[8 + 2 * j] = p.x; a[8 + 2 * j + 1] = p.y;
  }
  return a;
}

// B: 32x16 f16 where B[k][n] = Rows[n][k] (i.e. source stores columns as contiguous rows).
// Lane L holds column N=L%16; g=L/16; half-elems 0..7 -> K = 8g + {0..7}; 8..15 -> K = 16 + 8g + {0..7}.
__device__ __forceinline__ v16h load_b_frag_rows(const half_t* Rows, int ld, int n_base, int k_base) {
  int lane = threadIdx.x & 31;
  int g = lane >> 4;
  const half_t* p = Rows + (size_t)(n_base + (lane & 15)) * ld + k_base + 8 * g;
  v8h lo = *reinterpret_cast<const v8h*>(p);
  v8h hi = *reinterpret_cast<const v8h*>(p + 16);
  v16h b;
#pragma unroll
  for (int i = 0; i < 8; ++i) { b[i] = lo[i]; b[8 + i] = hi[i]; }
  return b;
}

__device__ __forceinline__ v8f wmma_f16(v16h a, v16h b, v8f c) {
  return __builtin_amdgcn_wmma_f32_16x16x32_f16(false, a, false, b, (short)0, c, false, false);
}

// ---------------- prep kernels ----------------
__global__ void k_cvt_f16(const float* __restrict__ src, half_t* __restrict__ dst, int n) {
  int i = blockIdx.x * blockDim.x + threadIdx.x;
  if (i < n) dst[i] = (half_t)src[i];
}

// W [G][Dd][N] f32  ->  WT [G][N][Dd] f16
__global__ void k_transpose_w(const float* __restrict__ W, half_t* __restrict__ WT,
                              int G, int Dd, int N) {
  int i = blockIdx.x * blockDim.x + threadIdx.x;
  if (i >= G * Dd * N) return;
  int n = i % N; int t = i / N; int d = t % Dd; int g = t / Dd;
  WT[((size_t)g * N + n) * Dd + d] = (half_t)W[i];
}

__global__ void k_zero_f32(float* __restrict__ p, int n) {
  int i = blockIdx.x * blockDim.x + threadIdx.x;
  if (i < n) p[i] = 0.f;
}

// ---------------- per-head projection: OUT[h] = X(4096x512) * W[h](512x64) ----------------
__global__ void k_proj_head(const half_t* __restrict__ X,   // [ROWS][DD] f16
                            const half_t* __restrict__ WT,  // [H][DKK][DD] f16 (transposed)
                            half_t* __restrict__ OUT) {     // [H][ROWS][DKK] f16
  int h = blockIdx.z;
  int mb = blockIdx.x * 16;
  int nb = blockIdx.y * 16;
  const half_t* wt = WT + (size_t)h * DKK * DD;
  half_t* out = OUT + (size_t)h * ROWS * DKK;
  v8f acc = {};
  for (int kb = 0; kb < DD; kb += 32) {
    v16h a = load_a_frag(X, DD, mb, kb);
    v16h bf = load_b_frag_rows(wt, DD, nb, kb);
    acc = wmma_f16(a, bf, acc);
  }
  int lane = threadIdx.x & 31;
  int n = lane & 15, g = lane >> 4;
#pragma unroll
  for (int r = 0; r < 8; ++r)
    out[(size_t)(mb + r + 8 * g) * DKK + nb + n] = (half_t)acc[r];
}

// V projection stored transposed: VT[b][k][s] so attn*V B-fragments are contiguous.
__global__ void k_proj_v(const half_t* __restrict__ X,   // [ROWS][DD] f16
                         const half_t* __restrict__ WT,  // [DKK][DD] f16
                         half_t* __restrict__ VT) {      // [BB][DKK][SS] f16
  int mb = blockIdx.x * 16;
  int nb = blockIdx.y * 16;
  v8f acc = {};
  for (int kb = 0; kb < DD; kb += 32) {
    v16h a = load_a_frag(X, DD, mb, kb);
    v16h bf = load_b_frag_rows(WT, DD, nb, kb);
    acc = wmma_f16(a, bf, acc);
  }
  int lane = threadIdx.x & 31;
  int n = lane & 15, g = lane >> 4;
#pragma unroll
  for (int r = 0; r < 8; ++r) {
    int t = mb + r + 8 * g;
    int b = t >> 11, s = t & (SS - 1);
    VT[((size_t)b * DKK + (nb + n)) * SS + s] = (half_t)acc[r];
  }
}

// ---------------- fused scores -> softmax -> attn write -> attn*V -> head-mean ----------------
__launch_bounds__(256, 1)
__global__ void k_attn(const half_t* __restrict__ QH,   // [H][ROWS][DKK]
                       const half_t* __restrict__ KH,   // [H][ROWS][DKK]
                       const half_t* __restrict__ VT,   // [BB][DKK][SS]
                       const float* __restrict__ scale_p,
                       float* __restrict__ attn_out,    // [H][BB][SS][SS]
                       float* __restrict__ ysum) {      // [BB][SS][DKK] f32 (pre-zeroed)
  extern __shared__ char smem[];
  float*  s_sc = (float*)smem;                     // 16 x 2048 f32 (128 KB)
  half_t* s_at = (half_t*)(smem + 16 * SS * 4);    // 16 x 2048 f16 (64 KB)

  int h = blockIdx.z, b = blockIdx.y, st = blockIdx.x;
  int wave = threadIdx.x >> 5, lane = threadIdx.x & 31;
  float scale = *scale_p;
  const half_t* qh = QH + (size_t)h * ROWS * DKK;
  const half_t* kh = KH + (size_t)h * ROWS * DKK;
  int m0 = b * SS + st * 16;
  int n = lane & 15, g = lane >> 4;

  // Q tile fragments (16 x 64) reused across all 128 key tiles
  v16h aq0 = load_a_frag(qh, DKK, m0, 0);
  v16h aq1 = load_a_frag(qh, DKK, m0, 32);

  for (int tt = wave; tt < SS / 16; tt += 8) {
    v16h b0 = load_b_frag_rows(kh, DKK, b * SS + tt * 16, 0);
    v16h b1 = load_b_frag_rows(kh, DKK, b * SS + tt * 16, 32);
    v8f acc = {};
    acc = wmma_f16(aq0, b0, acc);
    acc = wmma_f16(aq1, b1, acc);
#pragma unroll
    for (int r = 0; r < 8; ++r)
      s_sc[(r + 8 * g) * SS + tt * 16 + n] = acc[r] * scale;
  }
  __syncthreads();

  // Row softmax: wave w owns rows 2w, 2w+1. Write f32 probs to global, f16 copy to LDS.
#pragma unroll
  for (int rr = 0; rr < 2; ++rr) {
    int row = wave * 2 + rr;
    float* srow = s_sc + row * SS;
    float m = -3.402823466e38f;
    for (int c = lane; c < SS; c += 32) m = fmaxf(m, srow[c]);
#pragma unroll
    for (int off = 16; off > 0; off >>= 1) m = fmaxf(m, __shfl_xor(m, off, 32));
    float sum = 0.f;
    for (int c = lane; c < SS; c += 32) { float e = __expf(srow[c] - m); srow[c] = e; sum += e; }
#pragma unroll
    for (int off = 16; off > 0; off >>= 1) sum += __shfl_xor(sum, off, 32);
    float inv = 1.f / sum;
    float* grow = attn_out + (((size_t)h * BB + b) * SS + st * 16 + row) * SS;
    half_t* arow = s_at + row * SS;
    for (int c = lane; c < SS; c += 32) {
      float p = srow[c] * inv;
      grow[c] = p;
      arow[c] = (half_t)p;
    }
  }
  __syncthreads();

  // heads tile (16 x 64) = s_at(16x2048) * V(2048x64); 4 N-tiles x 2 K-halves across 8 waves
  const half_t* vt = VT + (size_t)b * DKK * SS;
  int ntile = wave >> 1, khalf = wave & 1;
  v8f acc = {};
  for (int kb = khalf * (SS / 2); kb < (khalf + 1) * (SS / 2); kb += 32) {
    v16h a = load_a_frag(s_at, SS, 0, kb);
    v16h bf = load_b_frag_rows(vt, SS, ntile * 16, kb);
    acc = wmma_f16(a, bf, acc);
  }
  float* part = s_sc + wave * 256;   // scores strip is dead now; reuse as reduction buffer
#pragma unroll
  for (int r = 0; r < 8; ++r) part[(r + 8 * g) * 16 + n] = acc[r];
  __syncthreads();
  if ((wave & 1) == 0) {
    float* p0 = s_sc + wave * 256;
    float* p1 = p0 + 256;
#pragma unroll
    for (int r = 0; r < 8; ++r) {
      float v = (p0[(r + 8 * g) * 16 + n] + p1[(r + 8 * g) * 16 + n]) * (1.0f / HH);
      atomicAdd(&ysum[((size_t)b * SS + st * 16 + r + 8 * g) * DKK + ntile * 16 + n], v);
    }
  }
}

// ---------------- final 64x64 output projection (tiny, VALU) ----------------
__global__ void k_outproj(const float* __restrict__ ysum, const float* __restrict__ Wo,
                          float* __restrict__ y) {
  int i = blockIdx.x * blockDim.x + threadIdx.x;   // ROWS*DKK
  int row = i >> 6, j = i & 63;
  float acc = 0.f;
#pragma unroll
  for (int k = 0; k < DKK; ++k) acc += ysum[(size_t)row * DKK + k] * Wo[k * DKK + j];
  y[i] = acc;
}

extern "C" void kernel_launch(void* const* d_in, const int* in_sizes, int n_in,
                              void* d_out, int out_size, void* d_ws, size_t ws_size,
                              hipStream_t stream) {
  (void)in_sizes; (void)n_in; (void)out_size; (void)ws_size;
  const float* q     = (const float*)d_in[0];
  const float* k     = (const float*)d_in[1];
  const float* v     = (const float*)d_in[2];
  const float* Wq    = (const float*)d_in[3];
  const float* Wk    = (const float*)d_in[4];
  const float* Wv    = (const float*)d_in[5];
  const float* Wo    = (const float*)d_in[6];
  const float* scale = (const float*)d_in[7];

  float* y_out    = (float*)d_out;
  float* attn_out = y_out + (size_t)ROWS * DKK;

  char* ws = (char*)d_ws;
  auto alloc = [&](size_t bytes) -> char* {
    char* p = ws; ws += (bytes + 255) & ~(size_t)255; return p;
  };
  half_t* q16  = (half_t*)alloc((size_t)ROWS * DD * 2);
  half_t* k16  = (half_t*)alloc((size_t)ROWS * DD * 2);
  half_t* v16  = (half_t*)alloc((size_t)ROWS * DD * 2);
  half_t* WqT  = (half_t*)alloc((size_t)HH * DKK * DD * 2);
  half_t* WkT  = (half_t*)alloc((size_t)HH * DKK * DD * 2);
  half_t* WvT  = (half_t*)alloc((size_t)DKK * DD * 2);
  half_t* qh   = (half_t*)alloc((size_t)HH * ROWS * DKK * 2);
  half_t* kh   = (half_t*)alloc((size_t)HH * ROWS * DKK * 2);
  half_t* vhT  = (half_t*)alloc((size_t)BB * DKK * SS * 2);
  float*  ysum = (float*)alloc((size_t)ROWS * DKK * 4);

  int n_act = ROWS * DD;
  k_cvt_f16<<<(n_act + 255) / 256, 256, 0, stream>>>(q, q16, n_act);
  k_cvt_f16<<<(n_act + 255) / 256, 256, 0, stream>>>(k, k16, n_act);
  k_cvt_f16<<<(n_act + 255) / 256, 256, 0, stream>>>(v, v16, n_act);

  int n_w = HH * DD * DKK;
  k_transpose_w<<<(n_w + 255) / 256, 256, 0, stream>>>(Wq, WqT, HH, DD, DKK);
  k_transpose_w<<<(n_w + 255) / 256, 256, 0, stream>>>(Wk, WkT, HH, DD, DKK);
  k_transpose_w<<<(DD * DKK + 255) / 256, 256, 0, stream>>>(Wv, WvT, 1, DD, DKK);

  k_zero_f32<<<(ROWS * DKK + 255) / 256, 256, 0, stream>>>(ysum, ROWS * DKK);

  dim3 pg(ROWS / 16, DKK / 16, HH);
  k_proj_head<<<pg, 32, 0, stream>>>(q16, WqT, qh);
  k_proj_head<<<pg, 32, 0, stream>>>(k16, WkT, kh);
  k_proj_v<<<dim3(ROWS / 16, DKK / 16), 32, 0, stream>>>(v16, WvT, vhT);

  hipFuncSetAttribute((const void*)k_attn,
                      hipFuncAttributeMaxDynamicSharedMemorySize, SMEM_BYTES);
  k_attn<<<dim3(SS / 16, BB, HH), 256, SMEM_BYTES, stream>>>(qh, kh, vhT, scale,
                                                            attn_out, ysum);

  k_outproj<<<(ROWS * DKK) / 256, 256, 0, stream>>>(ysum, Wo, y_out);
}
